// GNN_VCG_42047729827852
// MI455X (gfx1250) — matrix operand order repaired and checked by reading, not compiled, not run on hardware
//
#include <hip/hip_runtime.h>
#include <hip/hip_bf16.h>

#define DIM 128

typedef __attribute__((ext_vector_type(16))) __bf16 v16bf;
typedef __attribute__((ext_vector_type(8)))  __bf16 v8bf;
typedef __attribute__((ext_vector_type(8)))  float  v8f;

// ---------- helpers ----------

__device__ __forceinline__ __bf16 f2bf(float f) {
  union { float f; unsigned u; } x; x.f = f;
  unsigned r = x.u + 0x7FFFu + ((x.u >> 16) & 1u);   // round-to-nearest-even
  union { unsigned short s; __bf16 b; } y; y.s = (unsigned short)(r >> 16);
  return y.b;
}

__device__ __forceinline__ v8f wmma_bf16(v16bf a, v16bf b, v8f c) {
  // D = A(16x32 bf16) * B(32x16 bf16) + C(16x16 f32)
  return __builtin_amdgcn_wmma_f32_16x16x32_bf16(
      /*neg_a=*/false, a, /*neg_b=*/false, b,
      /*c_mod=*/(short)0, c, /*reuse_a=*/false, /*reuse_b=*/false);
}

// A-fragment (16x32, bf16) from fp32 row-major X, per ISA layout:
// lane&15 = M row; lane>>4 selects K sub-blocks {0-7,16-23} vs {8-15,24-31}
__device__ __forceinline__ v16bf load_a_f32(const float* __restrict__ row, int k0, int hi) {
  const float* p0 = row + k0 + hi * 8;
  const float* p1 = p0 + 16;
  v16bf a;
#pragma unroll
  for (int i = 0; i < 8; ++i) { a[i] = f2bf(p0[i]); a[8 + i] = f2bf(p1[i]); }
  return a;
}

// B-fragment (32x16, bf16) from transposed weights Wt[n][k] (row stride ldk):
// lanes 0-15: N=lane, K={k0..k0+7, k0+16..k0+23} ; lanes 16-31: N=lane-16, K={k0+8..k0+15, k0+24..k0+31}
__device__ __forceinline__ v16bf load_b_bf16(const __bf16* __restrict__ Wt, int ldk,
                                             int n, int k0, int hi) {
  const __bf16* p0 = Wt + (size_t)n * ldk + k0 + hi * 8;
  v8bf x0 = *(const v8bf*)p0;
  v8bf x1 = *(const v8bf*)(p0 + 16);
  v16bf b;
#pragma unroll
  for (int i = 0; i < 8; ++i) { b[i] = x0[i]; b[8 + i] = x1[i]; }
  return b;
}

// A-fragment from LDS bf16 tile (row stride 136 to dodge bank conflicts)
__device__ __forceinline__ v16bf load_a_lds(const __bf16* H, int m, int k0, int hi) {
  const __bf16* p0 = H + m * 136 + k0 + hi * 8;
  v8bf x0 = *(const v8bf*)p0;
  v8bf x1 = *(const v8bf*)(p0 + 16);
  v16bf a;
#pragma unroll
  for (int i = 0; i < 8; ++i) { a[i] = x0[i]; a[8 + i] = x1[i]; }
  return a;
}

// ---------- fused 2-layer MLP (Linear->ReLU->Linear), two MLPs per block ----------
// 8 waves: waves 0-3 -> MLP a, waves 4-7 -> MLP b; each wave owns a 16-row tile.
__global__ __launch_bounds__(256) void mlp_msg_kernel(
    const float* __restrict__ X, int nRows,
    const __bf16* __restrict__ W1ta, const __bf16* __restrict__ W2ta,
    const float* __restrict__ b1a, const float* __restrict__ b2a,
    const __bf16* __restrict__ W1tb, const __bf16* __restrict__ W2tb,
    const float* __restrict__ b1b, const float* __restrict__ b2b,
    float* __restrict__ Ya, float* __restrict__ Yb) {
  __shared__ __bf16 Hs[8 * 16 * 136];   // per-wave 16x128 bf16 ReLU tile (padded)

  const int wave = threadIdx.x >> 5;
  const int lane = threadIdx.x & 31;
  const int m    = lane & 15;
  const int hi   = lane >> 4;
  const int sel  = wave >> 2;
  const int mtile = blockIdx.x * 4 + (wave & 3);
  if (mtile * 16 >= nRows) return;      // wave-uniform: EXEC stays all-ones for WMMA
  const int m0 = mtile * 16;

  const __bf16* W1t = sel ? W1tb : W1ta;
  const __bf16* W2t = sel ? W2tb : W2ta;
  const float*  b1  = sel ? b1b  : b1a;
  const float*  b2  = sel ? b2b  : b2a;
  float*        Y   = sel ? Yb   : Ya;

  // ---- layer 1: H = relu(X @ W1 + b1) ----
  const float* xrow = X + (size_t)(m0 + m) * DIM;
  v16bf afr[4];
#pragma unroll
  for (int kc = 0; kc < 4; ++kc) afr[kc] = load_a_f32(xrow, kc * 32, hi);

  v8f acc[8];
#pragma unroll
  for (int nt = 0; nt < 8; ++nt) {
    float bv = b1[nt * 16 + m];
#pragma unroll
    for (int j = 0; j < 8; ++j) acc[nt][j] = bv;
  }
#pragma unroll
  for (int kc = 0; kc < 4; ++kc)
#pragma unroll
    for (int nt = 0; nt < 8; ++nt)
      acc[nt] = wmma_bf16(afr[kc], load_b_bf16(W1t, DIM, nt * 16 + m, kc * 32, hi), acc[nt]);

  __bf16* Hw = &Hs[wave * 16 * 136];
  // C/D layout: element j of lane -> (M = j + 8*hi, N = nt*16 + (lane&15))
#pragma unroll
  for (int nt = 0; nt < 8; ++nt)
#pragma unroll
    for (int j = 0; j < 8; ++j)
      Hw[(j + hi * 8) * 136 + nt * 16 + m] = f2bf(fmaxf(acc[nt][j], 0.0f));

  // ---- layer 2: Y = H @ W2 + b2  (wave-local LDS tile; DS in-order per wave) ----
  v16bf hfr[4];
#pragma unroll
  for (int kc = 0; kc < 4; ++kc) hfr[kc] = load_a_lds(Hw, m, kc * 32, hi);

  v8f acc2[8];
#pragma unroll
  for (int nt = 0; nt < 8; ++nt) {
    float bv = b2[nt * 16 + m];
#pragma unroll
    for (int j = 0; j < 8; ++j) acc2[nt][j] = bv;
  }
#pragma unroll
  for (int kc = 0; kc < 4; ++kc)
#pragma unroll
    for (int nt = 0; nt < 8; ++nt)
      acc2[nt] = wmma_bf16(hfr[kc], load_b_bf16(W2t, DIM, nt * 16 + m, kc * 32, hi), acc2[nt]);

#pragma unroll
  for (int nt = 0; nt < 8; ++nt)
#pragma unroll
    for (int j = 0; j < 8; ++j)
      Y[(size_t)(m0 + j + hi * 8) * DIM + nt * 16 + m] = acc2[nt][j];
}

// ---------- node update: Y = [emb | aggP | aggN](Nx384) @ Wt^T(384x128) + b ----------
__global__ __launch_bounds__(256) void upd_kernel(
    const float* __restrict__ emb, const float* __restrict__ aggP,
    const float* __restrict__ aggN, const __bf16* __restrict__ Wt,
    const float* __restrict__ bias, float* __restrict__ Y, int nRows) {
  const int wave = threadIdx.x >> 5;
  const int lane = threadIdx.x & 31;
  const int m  = lane & 15;
  const int hi = lane >> 4;
  const int mtile = blockIdx.x * 8 + wave;
  if (mtile * 16 >= nRows) return;
  const int m0 = mtile * 16;

  v8f acc[8];
#pragma unroll
  for (int nt = 0; nt < 8; ++nt) {
    float bv = bias[nt * 16 + m];
#pragma unroll
    for (int j = 0; j < 8; ++j) acc[nt][j] = bv;
  }

#pragma unroll
  for (int kc = 0; kc < 12; ++kc) {
    const float* src = (kc < 4) ? emb : (kc < 8) ? aggP : aggN;
    v16bf a = load_a_f32(src + (size_t)(m0 + m) * DIM, (kc & 3) * 32, hi);
#pragma unroll
    for (int nt = 0; nt < 8; ++nt)
      acc[nt] = wmma_bf16(a, load_b_bf16(Wt, 384, nt * 16 + m, kc * 32, hi), acc[nt]);
  }

#pragma unroll
  for (int nt = 0; nt < 8; ++nt)
#pragma unroll
    for (int j = 0; j < 8; ++j)
      Y[(size_t)(m0 + j + hi * 8) * DIM + nt * 16 + m] = acc[nt][j];
}

// ---------- edge scatter: aggr[dst[e]] += msg[src[e]] * rn[e] (one wave per edge) ----------
__global__ void k_scatter(const float* __restrict__ msg, const int* __restrict__ src,
                          const int* __restrict__ dst, const float* __restrict__ rn,
                          float* __restrict__ aggr, int nE) {
  int gid  = blockIdx.x * blockDim.x + threadIdx.x;
  int e    = gid >> 5;
  int lane = gid & 31;
  if (e >= nE) return;
  int s = src[e], d = dst[e];
  float r = rn[e];
  float4 mv = ((const float4*)(msg + (size_t)s * DIM))[lane];
  float* arow = aggr + (size_t)d * DIM + lane * 4;
  unsafeAtomicAdd(arow + 0, mv.x * r);
  unsafeAtomicAdd(arow + 1, mv.y * r);
  unsafeAtomicAdd(arow + 2, mv.z * r);
  unsafeAtomicAdd(arow + 3, mv.w * r);
}

// ---------- small utility kernels ----------
__global__ void k_zero(float* p, size_t n) {
  size_t i = blockIdx.x * (size_t)blockDim.x + threadIdx.x;
  size_t st = gridDim.x * (size_t)blockDim.x;
  for (; i < n; i += st) p[i] = 0.0f;
}
__global__ void k_copy(float* __restrict__ dst, const float* __restrict__ src, size_t n) {
  size_t i = blockIdx.x * (size_t)blockDim.x + threadIdx.x;
  size_t st = gridDim.x * (size_t)blockDim.x;
  for (; i < n; i += st) dst[i] = src[i];
}
__global__ void k_gather_idx(const int* __restrict__ attr, const int* __restrict__ sel,
                             int* __restrict__ outp, int n) {
  int i = blockIdx.x * blockDim.x + threadIdx.x;
  if (i < n) outp[i] = attr[sel[i]];
}
__global__ void k_deg(const int* __restrict__ idx, float* __restrict__ deg, int n) {
  int i = blockIdx.x * blockDim.x + threadIdx.x;
  if (i < n) unsafeAtomicAdd(&deg[idx[i]], 1.0f);
}
__global__ void k_rnorm(const int* __restrict__ a, const int* __restrict__ b,
                        const float* __restrict__ dA, const float* __restrict__ dB,
                        float* __restrict__ rn, int n) {
  int i = blockIdx.x * blockDim.x + threadIdx.x;
  if (i < n) rn[i] = rsqrtf(fmaxf(dA[a[i]], 1.0f) * fmaxf(dB[b[i]], 1.0f));
}
// mlp weights: W[mlp][k][n] (4x128x128 f32) -> Wt[mlp][n][k] bf16
__global__ void k_conv_mlp_w(const float* __restrict__ W, __bf16* __restrict__ Wt) {
  int t = blockIdx.x * blockDim.x + threadIdx.x;
  if (t < 4 * 128 * 128) {
    int mi = t >> 14, r = t & 16383, k = r >> 7, n = r & 127;
    Wt[(mi << 14) + (n << 7) + k] = f2bf(W[t]);
  }
}
// update weights: W[k][n] (384x128 f32) -> Wt[n][k] bf16 (ldk=384)
__global__ void k_conv_upd_w(const float* __restrict__ W, __bf16* __restrict__ Wt) {
  int t = blockIdx.x * blockDim.x + threadIdx.x;
  if (t < 384 * 128) {
    int k = t >> 7, n = t & 127;
    Wt[n * 384 + k] = f2bf(W[t]);
  }
}

// ---------- orchestration ----------
extern "C" void kernel_launch(void* const* d_in, const int* in_sizes, int n_in,
                              void* d_out, int out_size, void* d_ws, size_t ws_size,
                              hipStream_t stream) {
  (void)n_in; (void)out_size; (void)ws_size;
  const int EP = in_sizes[4];
  const int EN = in_sizes[5];
  const int V  = in_sizes[6] / DIM;
  const int C  = in_sizes[7] / DIM;

  const int*   v_edge = (const int*)d_in[2];
  const int*   c_edge = (const int*)d_in[3];
  const int*   p_sel  = (const int*)d_in[4];
  const int*   n_sel  = (const int*)d_in[5];
  const float* v0     = (const float*)d_in[6];
  const float* c0     = (const float*)d_in[7];
  const float* W1     = (const float*)d_in[8];
  const float* b1     = (const float*)d_in[9];
  const float* W2     = (const float*)d_in[10];
  const float* b2     = (const float*)d_in[11];
  const float* cW     = (const float*)d_in[12];
  const float* cb     = (const float*)d_in[13];
  const float* vW     = (const float*)d_in[14];
  const float* vb     = (const float*)d_in[15];
  float* out = (float*)d_out;

  // carve workspace
  char* base = (char*)d_ws;
  size_t off = 0;
  auto carve = [&](size_t bytes) -> void* {
    void* p = (void*)(base + off);
    off = (off + bytes + 255) & ~(size_t)255;
    return p;
  };
  __bf16* W1t  = (__bf16*)carve((size_t)4 * 128 * 128 * sizeof(__bf16));
  __bf16* W2t  = (__bf16*)carve((size_t)4 * 128 * 128 * sizeof(__bf16));
  __bf16* cWt  = (__bf16*)carve((size_t)128 * 384 * sizeof(__bf16));
  __bf16* vWt  = (__bf16*)carve((size_t)128 * 384 * sizeof(__bf16));
  int*    pv   = (int*)carve((size_t)EP * 4);
  int*    pc   = (int*)carve((size_t)EP * 4);
  int*    nv   = (int*)carve((size_t)EN * 4);
  int*    nc   = (int*)carve((size_t)EN * 4);
  float*  degs = (float*)carve((size_t)(2 * V + 2 * C) * 4); // [degPv|degNv|degPc|degNc]
  float*  degPv = degs, *degNv = degs + V, *degPc = degs + 2 * V, *degNc = degs + 2 * V + C;
  float*  prn  = (float*)carve((size_t)EP * 4);
  float*  nrn  = (float*)carve((size_t)EN * 4);
  float*  MSG  = (float*)carve((size_t)2 * C * DIM * 4);  // message buffer (reused v/c phase)
  float*  AGG  = (float*)carve((size_t)2 * C * DIM * 4);  // aggregate buffer (reused)

  const size_t vSlot = (size_t)V * DIM;
  const size_t cSlot = (size_t)C * DIM;
  float* outC = out + 5 * vSlot;   // c_embs stack starts after v_embs stack

  // ---- one-time setup ----
  k_conv_mlp_w<<<(4 * 128 * 128 + 255) / 256, 256, 0, stream>>>(W1, W1t);
  k_conv_mlp_w<<<(4 * 128 * 128 + 255) / 256, 256, 0, stream>>>(W2, W2t);
  k_conv_upd_w<<<(384 * 128 + 255) / 256, 256, 0, stream>>>(cW, cWt);
  k_conv_upd_w<<<(384 * 128 + 255) / 256, 256, 0, stream>>>(vW, vWt);
  k_gather_idx<<<(EP + 255) / 256, 256, 0, stream>>>(v_edge, p_sel, pv, EP);
  k_gather_idx<<<(EP + 255) / 256, 256, 0, stream>>>(c_edge, p_sel, pc, EP);
  k_gather_idx<<<(EN + 255) / 256, 256, 0, stream>>>(v_edge, n_sel, nv, EN);
  k_gather_idx<<<(EN + 255) / 256, 256, 0, stream>>>(c_edge, n_sel, nc, EN);
  k_zero<<<1024, 256, 0, stream>>>(degs, (size_t)(2 * V + 2 * C));
  k_deg<<<(EP + 255) / 256, 256, 0, stream>>>(pv, degPv, EP);
  k_deg<<<(EN + 255) / 256, 256, 0, stream>>>(nv, degNv, EN);
  k_deg<<<(EP + 255) / 256, 256, 0, stream>>>(pc, degPc, EP);
  k_deg<<<(EN + 255) / 256, 256, 0, stream>>>(nc, degNc, EN);
  k_rnorm<<<(EP + 255) / 256, 256, 0, stream>>>(pv, pc, degPv, degPc, prn, EP);
  k_rnorm<<<(EN + 255) / 256, 256, 0, stream>>>(nv, nc, degNv, degNc, nrn, EN);
  k_copy<<<2048, 256, 0, stream>>>(out, v0, vSlot);
  k_copy<<<2048, 256, 0, stream>>>(outC, c0, cSlot);

  const int vMlpBlocks = ((V / 16) + 3) / 4;
  const int cMlpBlocks = ((C / 16) + 3) / 4;
  const int vUpdBlocks = ((V / 16) + 7) / 8;
  const int cUpdBlocks = ((C / 16) + 7) / 8;

  for (int t = 0; t < 4; ++t) {
    const float* vprev = out  + (size_t)t * vSlot;
    float*       vnext = out  + (size_t)(t + 1) * vSlot;
    const float* cprev = outC + (size_t)t * cSlot;
    float*       cnext = outC + (size_t)(t + 1) * cSlot;

    // v -> c messages: MLP0 (p) and MLP1 (n) on v_emb(t)
    mlp_msg_kernel<<<vMlpBlocks, 256, 0, stream>>>(
        vprev, V,
        W1t + 0 * 16384, W2t + 0 * 16384, b1 + 0 * 128, b2 + 0 * 128,
        W1t + 1 * 16384, W2t + 1 * 16384, b1 + 1 * 128, b2 + 1 * 128,
        MSG, MSG + vSlot);
    k_zero<<<4096, 256, 0, stream>>>(AGG, 2 * cSlot);
    k_scatter<<<(EP * 32 + 255) / 256, 256, 0, stream>>>(MSG,         pv, pc, prn, AGG,         EP);
    k_scatter<<<(EN * 32 + 255) / 256, 256, 0, stream>>>(MSG + vSlot, nv, nc, nrn, AGG + cSlot, EN);
    upd_kernel<<<cUpdBlocks, 256, 0, stream>>>(cprev, AGG, AGG + cSlot, cWt, cb, cnext, C);

    // c -> v messages: MLP2 (p) and MLP3 (n) on c_emb(t)  (pre-update snapshot)
    mlp_msg_kernel<<<cMlpBlocks, 256, 0, stream>>>(
        cprev, C,
        W1t + 2 * 16384, W2t + 2 * 16384, b1 + 2 * 128, b2 + 2 * 128,
        W1t + 3 * 16384, W2t + 3 * 16384, b1 + 3 * 128, b2 + 3 * 128,
        MSG, MSG + cSlot);
    k_zero<<<4096, 256, 0, stream>>>(AGG, 2 * vSlot);
    k_scatter<<<(EP * 32 + 255) / 256, 256, 0, stream>>>(MSG,         pc, pv, prn, AGG,         EP);
    k_scatter<<<(EN * 32 + 255) / 256, 256, 0, stream>>>(MSG + cSlot, nc, nv, nrn, AGG + vSlot, EN);
    upd_kernel<<<vUpdBlocks, 256, 0, stream>>>(vprev, AGG, AGG + vSlot, vWt, vb, vnext, V);
  }
}
